// Model_33105607917917
// MI455X (gfx1250) — matrix operand order repaired and checked by reading, not compiled, not run on hardware
//
#include <hip/hip_runtime.h>
#include <hip/hip_bf16.h>
#include <stdint.h>

// ---------------------------------------------------------------------------
// Mamba-ish 4-layer model on MI455X (gfx1250).
// GEMMs (local mixer / in_proj / out_proj) run on v_wmma_f32_16x16x32_bf16.
// The block-shared B tile (64N x 64K bf16, 8KB) is staged in LDS with
// double-buffered global_load_async_to_lds_b128 (ASYNCcnt); each barrier
// round covers 8 back-to-back WMMAs with hoisted ds_load fragments.
// ---------------------------------------------------------------------------

#define D_MODEL 1024
#define N_LAYER 4
#define WINDOW  8
#define D_STATE 16
#define D_CONV  4
#define D_INNER 1024
#define DT_RANK 64
#define SEQ_L   2048
#define BATCH   4
#define MTOT    (BATCH * SEQ_L)          // 8192 rows
#define EPS     1e-5f

typedef __bf16 bf16_t;
typedef bf16_t v16bf __attribute__((ext_vector_type(16)));
typedef float  v8f   __attribute__((ext_vector_type(8)));

union BF16Frag { uint4 u[2]; v16bf v; };   // 32 bytes = 16 bf16 = 8 VGPRs

__device__ __forceinline__ unsigned short f2bf(float f) {
    unsigned u = __float_as_uint(f);
    unsigned r = u + 0x7FFFu + ((u >> 16) & 1u);   // round-to-nearest-even
    return (unsigned short)(r >> 16);
}

// Low 32 bits of a generic pointer to LDS == LDS byte offset (aperture map:
// LDS offset = low 32 address bits); this is what the async-DMA VDST operand wants.
__device__ __forceinline__ uint32_t lds_addr32(const void* p) {
    return (uint32_t)(uintptr_t)p;
}

// ------------------------------ f32 -> bf16 --------------------------------
__global__ __launch_bounds__(256) void cvt_bf16_kernel(
    const float* __restrict__ in, unsigned short* __restrict__ out, int n) {
    int i = blockIdx.x * 256 + threadIdx.x;
    if (i < n) out[i] = f2bf(in[i]);
}

// ------------------------------ LayerNorm ----------------------------------
__global__ __launch_bounds__(256) void layernorm_kernel(
    const float* __restrict__ x, const float* __restrict__ w,
    const float* __restrict__ b, unsigned short* __restrict__ out_bf,
    float* __restrict__ out_f) {
    __shared__ float s1[256], s2[256];
    const int tid = threadIdx.x;
    const size_t row = blockIdx.x;
    const float* xr = x + row * D_MODEL;
    float sum = 0.f, sq = 0.f;
    for (int i = tid; i < D_MODEL; i += 256) { float v = xr[i]; sum += v; sq += v * v; }
    s1[tid] = sum; s2[tid] = sq; __syncthreads();
    for (int off = 128; off > 0; off >>= 1) {
        if (tid < off) { s1[tid] += s1[tid + off]; s2[tid] += s2[tid + off]; }
        __syncthreads();
    }
    const float mu  = s1[0] * (1.f / D_MODEL);
    const float var = s2[0] * (1.f / D_MODEL) - mu * mu;
    const float inv = rsqrtf(var + EPS);
    for (int i = tid; i < D_MODEL; i += 256) {
        float y = (xr[i] - mu) * inv * w[i] + b[i];
        if (out_bf) out_bf[row * D_MODEL + i] = f2bf(y);
        if (out_f)  out_f [row * D_MODEL + i] = y;
    }
}

// ------------------------------ WMMA GEMM ----------------------------------
// C[M,N] (f32) = A[M,K](bf16) * W[N,K]^T(bf16) (+ res)
// Block = 128 threads = 4 waves; wave computes a 16(M) x 64(N) tile; the
// 64(N) x 64(K) B tile is shared by all 4 waves and staged in LDS with
// double-buffered global_load_async_to_lds_b128. 8 WMMAs per barrier round.
// GATHER: A rows come from the windowed mixer gather over xn[B,L,D_MODEL]:
//         A[m, w*1024 + d] = xn[b, l-7+w, d]  (zero when l-7+w < 0)
template <bool GATHER, bool ADDRES>
__global__ __launch_bounds__(128) void gemm_wmma_kernel(
    const unsigned short* __restrict__ A, const unsigned short* __restrict__ W,
    const float* __restrict__ res, float* __restrict__ C,
    int M, int N, int K) {
    __shared__ unsigned short sB[2][64][64];   // [buf][n][k] : 2 x 8KB

    const int tid  = threadIdx.x;
    const int wave = tid >> 5;
    const int lane = tid & 31;
    const int lo16 = lane & 15;
    const int half = lane >> 4;

    const int m0 = blockIdx.x * 64 + wave * 16;
    const int n0 = blockIdx.y * 64;
    const int mrow = m0 + lo16;           // A row for this lane
    const int gb = mrow >> 11;            // batch (L = 2048)
    const int gl = mrow & (SEQ_L - 1);    // position in sequence
    const int nk = K >> 6;                // K / 64

    // Fill sB[buf] with W[n0 .. n0+63][kb*64 .. +63]:
    // 512 x 16B async transfers, 4 per thread.
    auto stage = [&](int kb, int buf) {
#pragma unroll
        for (int i = 0; i < 4; ++i) {
            const int q   = tid + i * 128;   // 0..511
            const int row = q >> 3;          // n within tile
            const int seg = q & 7;           // 16B segment within 128B k-row
            const unsigned short* gp =
                W + ((size_t)(n0 + row) * K + (kb << 6) + seg * 8);
            const uint32_t lo = lds_addr32(&sB[buf][row][seg * 8]);
            asm volatile("global_load_async_to_lds_b128 %0, %1, off"
                         :: "v"(lo), "v"(gp) : "memory");
        }
    };

    v8f acc[4];
#pragma unroll
    for (int j = 0; j < 4; ++j)
#pragma unroll
        for (int r = 0; r < 8; ++r) acc[j][r] = 0.f;

    // prologue: fill buffer 0
    stage(0, 0);
    asm volatile("s_wait_asynccnt 0x0" ::: "memory");
    __syncthreads();

    int cur = 0;
    for (int ki = 0; ki < nk; ++ki) {
        // kick DMA for the next k-block; overlaps this iteration's 8 WMMAs
        if (ki + 1 < nk) stage(ki + 1, cur ^ 1);

#pragma unroll
        for (int ks = 0; ks < 2; ++ks) {
            const int k0 = (ki << 6) + (ks << 5);
            // ---- A operand: VGPR0-3 hold K = k0+half*8+0..7, VGPR4-7 hold +16
            BF16Frag a;
            if (GATHER) {
                const int w  = k0 >> 10;          // window tap (block stays in 1 tap)
                const int d  = k0 & (D_MODEL - 1);
                const int lp = gl + w - (WINDOW - 1);
                if (lp >= 0) {
                    const unsigned short* ap =
                        A + ((size_t)(gb * SEQ_L + lp) * D_MODEL + d + half * 8);
                    a.u[0] = *(const uint4*)(ap);
                    a.u[1] = *(const uint4*)(ap + 16);
                } else {
                    a.u[0] = make_uint4(0u, 0u, 0u, 0u);
                    a.u[1] = make_uint4(0u, 0u, 0u, 0u);
                }
            } else {
                const unsigned short* ap = A + ((size_t)mrow * K + k0 + half * 8);
                a.u[0] = *(const uint4*)(ap);
                a.u[1] = *(const uint4*)(ap + 16);
                if (k0 + 32 < K)   // global_prefetch_b8 of next A tile
                    __builtin_prefetch((const void*)(ap + 32), 0, 3);
            }

            // ---- hoist all 4 B fragments from LDS, then 4-deep WMMA issue.
            // lane lo16 = n within subtile, K = ks*32 + half*8 + 0..7, then +16.
            BF16Frag b[4];
#pragma unroll
            for (int j = 0; j < 4; ++j) {
                const unsigned short* bp =
                    &sB[cur][j * 16 + lo16][(ks << 5) + half * 8];
                b[j].u[0] = *(const uint4*)(bp);
                b[j].u[1] = *(const uint4*)(bp + 16);
            }
#pragma unroll
            for (int j = 0; j < 4; ++j)
                acc[j] = __builtin_amdgcn_wmma_f32_16x16x32_bf16(
                    false, a.v, false, b[j].v, (short)0, acc[j], false, false);
        }

        // drain own async copies, then all-waves barrier before buffer swap
        asm volatile("s_wait_asynccnt 0x0" ::: "memory");
        __syncthreads();
        cur ^= 1;
    }

    // ---- store: lane lo16 = N, half*8 + r = M within tile
#pragma unroll
    for (int j = 0; j < 4; ++j) {
        const int n = n0 + j * 16 + lo16;
#pragma unroll
        for (int r = 0; r < 8; ++r) {
            const int m = m0 + half * 8 + r;
            const size_t idx = (size_t)m * N + n;
            float v = acc[j][r];
            if (ADDRES) v += res[idx];
            C[idx] = v;
        }
    }
}

// ------------------------------ conv4 + SiLU -------------------------------
__global__ __launch_bounds__(256) void conv_silu_kernel(
    const float* __restrict__ xz, const float* __restrict__ cw,
    const float* __restrict__ cb, float* __restrict__ xconv) {
    const int idx = blockIdx.x * 256 + threadIdx.x;
    if (idx >= MTOT * D_INNER) return;
    const int d = idx & (D_INNER - 1);
    const int m = idx >> 10;
    const int l = m & (SEQ_L - 1);
    const int b = m >> 11;
    float acc = cb[d];
#pragma unroll
    for (int i = 0; i < D_CONV; ++i) {
        const int lp = l - (D_CONV - 1) + i;
        if (lp >= 0)
            acc += xz[(size_t)(b * SEQ_L + lp) * (2 * D_INNER) + d] * cw[d * D_CONV + i];
    }
    xconv[idx] = acc / (1.f + expf(-acc));
}

// ------------------------------ x_proj -------------------------------------
__global__ __launch_bounds__(128) void xproj_kernel(
    const float* __restrict__ xconv, const float* __restrict__ Wx,
    float* __restrict__ xdbl) {
    __shared__ float srow[D_INNER];
    const size_t m = blockIdx.x;
    for (int i = threadIdx.x; i < D_INNER; i += 128)
        srow[i] = xconv[m * D_INNER + i];
    __syncthreads();
    const int j = threadIdx.x;
    if (j < DT_RANK + 2 * D_STATE) {
        const float* wr = Wx + (size_t)j * D_INNER;
        float acc = 0.f;
        for (int k = 0; k < D_INNER; ++k) acc += srow[k] * wr[k];
        xdbl[m * (DT_RANK + 2 * D_STATE) + j] = acc;
    }
}

// ------------------------------ dt_proj + softplus -------------------------
__global__ __launch_bounds__(256) void dtproj_kernel(
    const float* __restrict__ xdbl, const float* __restrict__ Wd,
    const float* __restrict__ bd, float* __restrict__ delta) {
    const int t = blockIdx.x * 256 + threadIdx.x;
    if (t >= MTOT * D_INNER) return;
    const int d = t & (D_INNER - 1);
    const int m = t >> 10;
    const float* dt = xdbl + (size_t)m * (DT_RANK + 2 * D_STATE);
    const float* wr = Wd + (size_t)d * DT_RANK;
    float acc = bd[d];
#pragma unroll 8
    for (int r = 0; r < DT_RANK; ++r) acc += dt[r] * wr[r];
    delta[t] = (acc > 20.f) ? acc : log1pf(expf(acc));   // softplus
}

// ------------------------------ selective scan + gate ----------------------
__global__ __launch_bounds__(256) void scan_kernel(
    const float* __restrict__ delta, const float* __restrict__ xconv,
    const float* __restrict__ xdbl, const float* __restrict__ A_log,
    const float* __restrict__ Dp, const float* __restrict__ xz,
    unsigned short* __restrict__ ygbf) {
    const int t = blockIdx.x * 256 + threadIdx.x;
    if (t >= BATCH * D_INNER) return;
    const int d = t & (D_INNER - 1);
    const int b = t >> 10;
    float A[D_STATE], h[D_STATE];
#pragma unroll
    for (int n = 0; n < D_STATE; ++n) {
        A[n] = -expf(A_log[d * D_STATE + n]);
        h[n] = 0.f;
    }
    const float Dd = Dp[d];
    for (int l = 0; l < SEQ_L; ++l) {
        const size_t m = (size_t)b * SEQ_L + l;
        const float dl = delta[m * D_INNER + d];
        const float u  = xconv[m * D_INNER + d];
        const float* row = xdbl + m * (DT_RANK + 2 * D_STATE);
        float y = 0.f;
#pragma unroll
        for (int n = 0; n < D_STATE; ++n) {
            const float dA = expf(dl * A[n]);
            h[n] = dA * h[n] + dl * row[DT_RANK + n] * u;
            y += h[n] * row[DT_RANK + D_STATE + n];
        }
        y += u * Dd;
        const float z = xz[m * (2 * D_INNER) + D_INNER + d];
        const float g = z / (1.f + expf(-z));
        ygbf[m * D_INNER + d] = f2bf(y * g);
    }
}

// ---------------------------------------------------------------------------
extern "C" void kernel_launch(void* const* d_in, const int* in_sizes, int n_in,
                              void* d_out, int out_size, void* d_ws, size_t ws_size,
                              hipStream_t stream) {
    // Input order (setup_inputs insertion order):
    // 0 hidden_states, 1 ln1_w, 2 ln1_b, 3 ln2_w, 4 ln2_b, 5 lnf_w, 6 lnf_b,
    // 7..10 local_ws[0..3],
    // 11+9l..: mamba_ps[l]: in_proj_w, conv_w, conv_b, x_proj_w, dt_proj_w,
    //          dt_proj_b, A_log, D, out_proj_w
    const float* hidden = (const float*)d_in[0];
    const float* ln1w = (const float*)d_in[1];
    const float* ln1b = (const float*)d_in[2];
    const float* ln2w = (const float*)d_in[3];
    const float* ln2b = (const float*)d_in[4];
    const float* lnfw = (const float*)d_in[5];
    const float* lnfb = (const float*)d_in[6];

    char* ws = (char*)d_ws;
    size_t off = 0;
    auto carve = [&](size_t bytes) -> char* {
        char* p = ws + off;
        off = (off + bytes + 255) & ~(size_t)255;
        return p;
    };
    float*          h      = (float*)         carve((size_t)MTOT * D_MODEL * 4);
    unsigned short* xnbf   = (unsigned short*)carve((size_t)MTOT * D_MODEL * 2);
    unsigned short* wlocbf = (unsigned short*)carve((size_t)D_MODEL * WINDOW * D_MODEL * 2);
    unsigned short* wipbf  = (unsigned short*)carve((size_t)2 * D_INNER * D_MODEL * 2);
    unsigned short* wopbf  = (unsigned short*)carve((size_t)D_MODEL * D_INNER * 2);
    float*          xz     = (float*)         carve((size_t)MTOT * 2 * D_INNER * 4);
    float*          xconv  = (float*)         carve((size_t)MTOT * D_INNER * 4);
    float*          xdbl   = (float*)         carve((size_t)MTOT * (DT_RANK + 2 * D_STATE) * 4);
    float*          delta  = (float*)         carve((size_t)MTOT * D_INNER * 4);
    unsigned short* ygbf   = (unsigned short*)carve((size_t)MTOT * D_INNER * 2);
    (void)ws_size;

    const dim3 blk128(128), blk256(256);

    for (int l = 0; l < N_LAYER; ++l) {
        const float* wloc = (const float*)d_in[7 + l];
        const int pb = 11 + 9 * l;
        const float* in_proj_w  = (const float*)d_in[pb + 0];
        const float* conv_w     = (const float*)d_in[pb + 1];
        const float* conv_b     = (const float*)d_in[pb + 2];
        const float* x_proj_w   = (const float*)d_in[pb + 3];
        const float* dt_proj_w  = (const float*)d_in[pb + 4];
        const float* dt_proj_b  = (const float*)d_in[pb + 5];
        const float* A_log      = (const float*)d_in[pb + 6];
        const float* Dp         = (const float*)d_in[pb + 7];
        const float* out_proj_w = (const float*)d_in[pb + 8];

        const float* hin = (l == 0) ? hidden : h;

        // ---- local mixer branch: h = hin + gather(LN1(hin)) @ Wl^T
        layernorm_kernel<<<MTOT, blk256, 0, stream>>>(hin, ln1w, ln1b, xnbf, nullptr);
        {
            const int n = D_MODEL * WINDOW * D_MODEL;
            cvt_bf16_kernel<<<(n + 255) / 256, blk256, 0, stream>>>(wloc, wlocbf, n);
        }
        gemm_wmma_kernel<true, true><<<dim3(MTOT / 64, D_MODEL / 64), blk128, 0, stream>>>(
            xnbf, wlocbf, hin, h, MTOT, D_MODEL, WINDOW * D_MODEL);

        // ---- mamba branch
        layernorm_kernel<<<MTOT, blk256, 0, stream>>>(h, ln2w, ln2b, xnbf, nullptr);
        {
            const int n = 2 * D_INNER * D_MODEL;
            cvt_bf16_kernel<<<(n + 255) / 256, blk256, 0, stream>>>(in_proj_w, wipbf, n);
        }
        gemm_wmma_kernel<false, false><<<dim3(MTOT / 64, (2 * D_INNER) / 64), blk128, 0, stream>>>(
            xnbf, wipbf, nullptr, xz, MTOT, 2 * D_INNER, D_MODEL);

        conv_silu_kernel<<<(MTOT * D_INNER) / 256, blk256, 0, stream>>>(xz, conv_w, conv_b, xconv);
        xproj_kernel<<<MTOT, blk128, 0, stream>>>(xconv, x_proj_w, xdbl);
        dtproj_kernel<<<(MTOT * D_INNER) / 256, blk256, 0, stream>>>(xdbl, dt_proj_w, dt_proj_b, delta);
        scan_kernel<<<(BATCH * D_INNER) / 256, blk256, 0, stream>>>(
            delta, xconv, xdbl, A_log, Dp, xz, ygbf);

        {
            const int n = D_MODEL * D_INNER;
            cvt_bf16_kernel<<<(n + 255) / 256, blk256, 0, stream>>>(out_proj_w, wopbf, n);
        }
        gemm_wmma_kernel<false, true><<<dim3(MTOT / 64, D_MODEL / 64), blk128, 0, stream>>>(
            ygbf, wopbf, h, h, MTOT, D_MODEL, D_INNER);
    }

    layernorm_kernel<<<MTOT, blk256, 0, stream>>>(h, lnfw, lnfb, nullptr, (float*)d_out);
    (void)in_sizes; (void)n_in; (void)out_size;
}
